// Block_32770600468774
// MI455X (gfx1250) — matrix operand order repaired and checked by reading, not compiled, not run on hardware
//
#include <hip/hip_runtime.h>
#include <math.h>

// ---------------- types for CDNA5 WMMA ----------------
typedef __bf16 v16bf __attribute__((ext_vector_type(16)));
typedef float  v8f   __attribute__((ext_vector_type(8)));
typedef int    v4i128 __attribute__((vector_size(4 * sizeof(int))));
typedef unsigned int u32x4 __attribute__((vector_size(4 * sizeof(int))));
typedef int    i32x4 __attribute__((vector_size(4 * sizeof(int))));
typedef int    i32x8 __attribute__((vector_size(8 * sizeof(int))));

union Frag16 { v16bf v; unsigned u[8]; unsigned short s[16]; };

static __device__ __forceinline__ unsigned short f2bf(float f) {
    unsigned u = __float_as_uint(f);
    unsigned r = u + 0x7FFFu + ((u >> 16) & 1u);   // RNE
    return (unsigned short)(r >> 16);
}
static __device__ __forceinline__ float bf2f(unsigned short b) {
    return __uint_as_float(((unsigned)b) << 16);
}

// ---------------- CDNA5 async / TDM staging with safe fallbacks ---------------
#if defined(__has_builtin)
#if __has_builtin(__builtin_amdgcn_global_load_async_to_lds_b128)
#define HAVE_ASYNC_LDS 1
#endif
#if __has_builtin(__builtin_amdgcn_tensor_load_to_lds)
#define HAVE_TDM 1
#endif
#endif

#ifdef HAVE_ASYNC_LDS
#define WAITA0 asm volatile("s_wait_asynccnt 0x0" ::: "memory")
#define WAITA2 asm volatile("s_wait_asynccnt 0x2" ::: "memory")
#define WAITA4 asm volatile("s_wait_asynccnt 0x4" ::: "memory")
#else
#define WAITA0
#define WAITA2
#define WAITA4
#endif

#ifdef HAVE_TDM
#define TWAIT1 __builtin_amdgcn_s_wait_tensorcnt(1)
#define TWAIT0 __builtin_amdgcn_s_wait_tensorcnt(0)
#define WAITA_NEXT WAITA2
#else
#define TWAIT1
#define TWAIT0
#define WAITA_NEXT WAITA4
#endif

static __device__ __forceinline__ void cp16(const unsigned short* g, unsigned short* l) {
#ifdef HAVE_ASYNC_LDS
    __builtin_amdgcn_global_load_async_to_lds_b128(
        (__attribute__((address_space(1))) v4i128*)(unsigned short*)g,
        (__attribute__((address_space(3))) v4i128*)l, 0, 0);
#else
    *(uint4*)l = *(const uint4*)g;
#endif
}

// ---------------- BF16 x BF16 WMMA GEMM, double-buffered LDS pipeline ----------
// A: bf16 [M][K] ; Bt: bf16 [Npad][K] (row-major, K contiguous, Npad mult of 64,
// rows >= N may contain garbage: they only feed discarded outputs). K % 32 == 0.
// A-tile staged via per-lane async-to-LDS (ASYNCcnt); B-tile staged via one
// Tensor Data Mover descriptor per step (TENSORcnt, wave 0 issues).
// mode 0: C = acc ; 1: C = acc + res ; 2: C += acc * (rowScale ? rowScale[m*ss] : 1)
__global__ __launch_bounds__(128) void gemm_bf16_wmma(
    const unsigned short* __restrict__ A, const unsigned short* __restrict__ Bt,
    float* __restrict__ C, const float* __restrict__ res,
    const float* __restrict__ rowScale, int scaleStride,
    int M, int N, int K, int mode)
{
    __shared__ alignas(16) unsigned short As[2][64][40];   // 64B data + 16B pad rows
    __shared__ alignas(16) unsigned short Bs[2][64][40];
    const int tid  = threadIdx.x;
    const int lane = tid & 31;
    const int wave = tid >> 5;
    const int lh = lane >> 4, ll = lane & 15;
    const int n0 = blockIdx.x * 64;
    const int m0 = blockIdx.y * 64;
    const int waveM = (wave >> 1) * 32;
    const int waveN = (wave & 1) * 32;

    const unsigned short* Abase = A  + (size_t)m0 * K;
    const unsigned short* Bbase = Bt + (size_t)n0 * K;

    v8f c[2][2];
    for (int i = 0; i < 2; ++i)
        for (int j = 0; j < 2; ++j)
            for (int e = 0; e < 8; ++e) c[i][j][e] = 0.0f;

    // A tile: 64x32 bf16, 256 x 16B chunks, 2 async transfers per thread
    auto stageA = [&](int p, int ko) {
        for (int i = 0; i < 2; ++i) {
            int cc = tid + i * 128;
            int row = cc >> 2, seg = cc & 3;
            cp16(Abase + (size_t)row * K + ko + seg * 8, &As[p][row][seg * 8]);
        }
    };
    // B tile: one TDM 2D descriptor (tile 32 elem x 64 rows, stride K, 2B data,
    // LDS rows padded 64B->80B via pad_interval=16DW, pad_amount=4DW)
    auto stageB = [&](int p, int ko) {
#ifdef HAVE_TDM
        if (wave == 0) {
            unsigned long long ga = (unsigned long long)(const void*)(Bbase + ko);
            unsigned ldsa = (unsigned)(unsigned long long)
                (__attribute__((address_space(3))) unsigned short*)&Bs[p][0][0];
            u32x4 g0;
            g0[0] = 1u;                                        // count=1 (valid user D#)
            g0[1] = ldsa;                                      // lds_addr (bytes)
            g0[2] = (unsigned)ga;                              // global_addr[31:0]
            g0[3] = (unsigned)((ga >> 32) & 0x1FFFFFFu) | (2u << 30);  // addr[56:32]|type=2
            i32x8 g1;
            g1[0] = (int)((1u << 16) | (1u << 20) | (3u << 22) | (3u << 25));
            //        data_size=2B    pad_enable   pad_int=16DW  pad_amt=4DW
            g1[1] = (int)(32u << 16);      // tensor_dim0 = 32 (tile==tensor view)
            g1[2] = (int)(64u << 16);      // tensor_dim1 = 64
            g1[3] = (int)(32u << 16);      // tile_dim0 = 32
            g1[4] = 64;                    // tile_dim1 = 64, tile_dim2 = 0
            g1[5] = (int)(unsigned)K;      // tensor_dim0_stride (elements)
            g1[6] = 0;
            g1[7] = 0;
            i32x4 z; z[0] = 0; z[1] = 0; z[2] = 0; z[3] = 0;   // 2D: groups 2/3 unused
            i32x8 z8; for (int q = 0; q < 8; ++q) z8[q] = 0;
            __builtin_amdgcn_tensor_load_to_lds(g0, g1, z, z, z8, 0);
        }
#else
        for (int i = 0; i < 2; ++i) {
            int cc = tid + i * 128;
            int row = cc >> 2, seg = cc & 3;
            cp16(Bbase + (size_t)row * K + ko + seg * 8, &Bs[p][row][seg * 8]);
        }
#endif
    };

    const int nk = K >> 5;
    stageA(0, 0);
    stageB(0, 0);
    for (int t = 0; t < nk; ++t) {
        const int cur = t & 1;
        if (t + 1 < nk) {
            stageA(cur ^ 1, (t + 1) << 5);
            stageB(cur ^ 1, (t + 1) << 5);
            WAITA_NEXT; TWAIT1;
        } else {
            WAITA0; TWAIT0;
        }
        __syncthreads();

        Frag16 a[2], b[2];
        for (int mt = 0; mt < 2; ++mt) {
            const unsigned* row32 = (const unsigned*)&As[cur][waveM + mt * 16 + ll][0];
            for (int i = 0; i < 8; ++i)
                a[mt].u[i] = row32[lh * 8 + i];
        }
        for (int nt = 0; nt < 2; ++nt) {
            const unsigned* row32 = (const unsigned*)&Bs[cur][waveN + nt * 16 + ll][0];
            for (int i = 0; i < 8; ++i) b[nt].u[i] = row32[lh * 8 + i];
        }
        for (int mt = 0; mt < 2; ++mt)
            for (int nt = 0; nt < 2; ++nt)
                c[mt][nt] = __builtin_amdgcn_wmma_f32_16x16x32_bf16(
                    false, a[mt].v, false, b[nt].v, (short)0, c[mt][nt], false, false);
        __syncthreads();
    }

    for (int mt = 0; mt < 2; ++mt)
        for (int nt = 0; nt < 2; ++nt)
            for (int r = 0; r < 8; ++r) {
                int gm = m0 + waveM + mt * 16 + lh * 8 + r;
                int gn = n0 + waveN + nt * 16 + ll;
                if (gm < M && gn < N) {
                    size_t off = (size_t)gm * N + gn;
                    float acc = c[mt][nt][r];
                    if (mode == 0)      C[off] = acc;
                    else if (mode == 1) C[off] = acc + res[off];
                    else {
                        float sc = rowScale ? rowScale[(size_t)gm * scaleStride] : 1.0f;
                        C[off] += acc * sc;
                    }
                }
            }
}

// ---------------- weight conversion (once per launch) ----------------
__global__ void conv_kn2nk_kernel(const float* __restrict__ in, unsigned short* __restrict__ out,
                                  int K, int N, int Npad) {
    int idx = blockIdx.x * 256 + threadIdx.x;
    if (idx >= Npad * K) return;
    int n = idx / K, k = idx - n * K;
    out[idx] = (n < N) ? f2bf(in[(size_t)k * N + n]) : (unsigned short)0;
}
__global__ void conv_nk_kernel(const float* __restrict__ in, unsigned short* __restrict__ out, int total) {
    int idx = blockIdx.x * 256 + threadIdx.x;
    if (idx < total) out[idx] = f2bf(in[idx]);
}

// ---------------- LayerNorm (f32 in, bf16 out) ----------------
__global__ __launch_bounds__(256) void ln_bf16_kernel(
    const float* __restrict__ in, int inStride,
    const float* __restrict__ w, const float* __restrict__ bias,
    unsigned short* __restrict__ out, int outStride, int n)
{
    __shared__ float red[256];
    int row = blockIdx.x;
    const float* x = in + (size_t)row * inStride;
    float s = 0.f;
    for (int i = threadIdx.x; i < n; i += 256) s += x[i];
    red[threadIdx.x] = s; __syncthreads();
    for (int st = 128; st > 0; st >>= 1) {
        if (threadIdx.x < st) red[threadIdx.x] += red[threadIdx.x + st];
        __syncthreads();
    }
    float mean = red[0] / n; __syncthreads();
    float vs = 0.f;
    for (int i = threadIdx.x; i < n; i += 256) { float d = x[i] - mean; vs += d * d; }
    red[threadIdx.x] = vs; __syncthreads();
    for (int st = 128; st > 0; st >>= 1) {
        if (threadIdx.x < st) red[threadIdx.x] += red[threadIdx.x + st];
        __syncthreads();
    }
    float rstd = rsqrtf(red[0] / n + 1e-5f);
    unsigned short* y = out + (size_t)row * outStride;
    for (int i = threadIdx.x; i < n; i += 256) {
        float v = (x[i] - mean) * rstd * w[i];
        if (bias) v += bias[i];
        y[i] = f2bf(v);
    }
}

// ---------------- RoPE on Q, fold 1/sqrt(64), f32 -> bf16 [T,H,64] ------------
__global__ __launch_bounds__(768) void rope_q_kernel(const float* __restrict__ Q,
                                                     unsigned short* __restrict__ qb)
{
    int t = blockIdx.x;
    int s = t & 1023;              // S = 1024
    int hh = threadIdx.x >> 6;
    int d  = threadIdx.x & 63;
    const float* qrow = Q + (size_t)t * 768 + hh * 64;
    float val;
    if (d < 32) val = qrow[d];
    else {
        int j = d - 32;
        float x1 = qrow[32 + j];
        float x2 = qrow[32 + ((j < 16) ? j + 16 : j - 16)];
        float ang = (float)s * powf(10000.f, -(float)(j & 15) / 32.f);
        float sn, cs; sincosf(ang, &sn, &cs);
        val = x1 * cs + ((j < 16) ? -x2 : x2) * sn;
    }
    qb[(size_t)t * 768 + hh * 64 + d] = f2bf(val * 0.125f);
}

// ---------------- assemble K (bf16 [T,H,64]) and V-transposed (bf16 [B,H,64,S])
__global__ __launch_bounds__(768) void assemble_kv_kernel(
    const float* __restrict__ KV, const float* __restrict__ ckv,
    unsigned short* __restrict__ kb, unsigned short* __restrict__ vT)
{
    int t = blockIdx.x;
    int b = t >> 10, s = t & 1023;
    int hh = threadIdx.x >> 6;
    int d  = threadIdx.x & 63;
    vT[(((size_t)b * 12 + hh) * 64 + d) * 1024 + s] = f2bf(KV[(size_t)t * 1152 + hh * 96 + 32 + d]);
    float kval;
    if (d < 32) {
        kval = KV[(size_t)t * 1152 + hh * 96 + d];
    } else {
        int j = d - 32;
        float x1 = ckv[(size_t)t * 544 + 512 + j];
        float x2 = ckv[(size_t)t * 544 + 512 + ((j < 16) ? j + 16 : j - 16)];
        float ang = (float)s * powf(10000.f, -(float)(j & 15) / 32.f);
        float sn, cs; sincosf(ang, &sn, &cs);
        kval = x1 * cs + ((j < 16) ? -x2 : x2) * sn;
    }
    kb[(size_t)t * 768 + hh * 64 + d] = f2bf(kval);
}

// ---------------- causal flash attention, WMMA QK^T and P.V -------------------
__global__ __launch_bounds__(128) void attn_kernel(
    const unsigned short* __restrict__ q, const unsigned short* __restrict__ k,
    const unsigned short* __restrict__ vT, unsigned short* __restrict__ o)
{
    __shared__ alignas(16) unsigned short Ps[4][16][34];
    const int tid = threadIdx.x, lane = tid & 31, wave = tid >> 5;
    const int lh = lane >> 4, ll = lane & 15;
    const int h = blockIdx.y, b = blockIdx.z;
    const int q0 = blockIdx.x * 64 + wave * 16;
    const int S = 1024, H = 12;

    Frag16 aq[2];
    {
        const unsigned* qrow = (const unsigned*)(q + (((size_t)(b * S + q0 + ll) * H) + h) * 64);
        for (int f = 0; f < 2; ++f)
            for (int i = 0; i < 8; ++i)
                aq[f].u[i] = qrow[f * 16 + lh * 8 + i];
    }
    v8f oacc[4];
    for (int nt = 0; nt < 4; ++nt) for (int e = 0; e < 8; ++e) oacc[nt][e] = 0.f;
    float mrow[8], lrow[8];
    for (int r = 0; r < 8; ++r) { mrow[r] = -1e30f; lrow[r] = 0.f; }

    const int kbmax = (q0 + 15) >> 5;
    for (int kb = 0; kb <= kbmax; ++kb) {
        const int kbase = kb * 32;
        v8f sc[2];
        for (int t = 0; t < 2; ++t) {
            Frag16 bk[2];
            const unsigned* krow = (const unsigned*)(k + (((size_t)(b * S + kbase + t * 16 + ll) * H) + h) * 64);
            for (int f = 0; f < 2; ++f)
                for (int i = 0; i < 8; ++i)
                    bk[f].u[i] = krow[f * 16 + lh * 8 + i];
            v8f sacc; for (int e = 0; e < 8; ++e) sacc[e] = 0.f;
            sacc = __builtin_amdgcn_wmma_f32_16x16x32_bf16(false, aq[0].v, false, bk[0].v, (short)0, sacc, false, false);
            sacc = __builtin_amdgcn_wmma_f32_16x16x32_bf16(false, aq[1].v, false, bk[1].v, (short)0, sacc, false, false);
            sc[t] = sacc;
        }
        for (int r = 0; r < 8; ++r) {
            int row = q0 + lh * 8 + r;
            int c0 = kbase + ll, c1 = kbase + 16 + ll;
            float s0 = (c0 <= row) ? sc[0][r] : -1e30f;
            float s1 = (c1 <= row) ? sc[1][r] : -1e30f;
            float mcur = fmaxf(s0, s1);
            for (int off = 1; off < 16; off <<= 1)
                mcur = fmaxf(mcur, __shfl_xor(mcur, off, 32));
            float mnew = fmaxf(mrow[r], mcur);
            float corr = expf(mrow[r] - mnew);
            float p0 = expf(s0 - mnew);
            float p1 = expf(s1 - mnew);
            float ls = p0 + p1;
            for (int off = 1; off < 16; off <<= 1)
                ls += __shfl_xor(ls, off, 32);
            lrow[r] = lrow[r] * corr + ls;
            mrow[r] = mnew;
            for (int nt = 0; nt < 4; ++nt) oacc[nt][r] *= corr;
            Ps[wave][lh * 8 + r][ll]      = f2bf(p0);
            Ps[wave][lh * 8 + r][16 + ll] = f2bf(p1);
        }
        asm volatile("s_wait_dscnt 0x0" ::: "memory");   // wave-local LDS RAW fence
        Frag16 pa;
        {
            const unsigned* prow = (const unsigned*)&Ps[wave][ll][0];
            for (int i = 0; i < 8; ++i)
                pa.u[i] = prow[lh * 8 + i];
        }
        for (int nt = 0; nt < 4; ++nt) {
            Frag16 bv;
            const unsigned* vrow = (const unsigned*)(vT + ((((size_t)b * H + h) * 64) + nt * 16 + ll) * S + kbase + lh * 16);
            for (int i = 0; i < 8; ++i) bv.u[i] = vrow[i];
            oacc[nt] = __builtin_amdgcn_wmma_f32_16x16x32_bf16(false, pa.v, false, bv.v, (short)0, oacc[nt], false, false);
        }
    }
    for (int nt = 0; nt < 4; ++nt)
        for (int r = 0; r < 8; ++r) {
            int row = q0 + lh * 8 + r;
            int d = nt * 16 + ll;
            o[(((size_t)(b * S + row) * H) + h) * 64 + d] = f2bf(oacc[nt][r] / lrow[r]);
        }
}

// ---------------- routing: sigmoid gates over top-2 of raw+bias ---------------
__global__ __launch_bounds__(128) void route_kernel(
    const unsigned short* __restrict__ h2, const float* __restrict__ cent,
    const float* __restrict__ rbias, float* __restrict__ gates)
{
    int lane = threadIdx.x & 31, wave = threadIdx.x >> 5;
    int t = blockIdx.x * 4 + wave;
    float raw[8];
    for (int e = 0; e < 8; ++e) {
        float a = 0.f;
        for (int dd = lane; dd < 768; dd += 32)
            a += bf2f(h2[(size_t)t * 768 + dd]) * cent[(size_t)e * 768 + dd];
        for (int off = 16; off > 0; off >>= 1) a += __shfl_xor(a, off, 32);
        raw[e] = a;
    }
    float bal[8];
    for (int e = 0; e < 8; ++e) bal[e] = raw[e] + rbias[e];
    int i1 = 0; float b1 = bal[0];
    for (int e = 1; e < 8; ++e) if (bal[e] > b1) { b1 = bal[e]; i1 = e; }
    int i2 = -1; float b2 = -1e30f;
    for (int e = 0; e < 8; ++e) if (e != i1 && bal[e] > b2) { b2 = bal[e]; i2 = e; }
    float w1 = 1.f / (1.f + expf(-raw[i1]));
    float w2 = 1.f / (1.f + expf(-raw[i2]));
    float inv = 1.f / (w1 + w2 + 1e-9f);
    if (lane < 8) {
        float g = (lane == i1) ? w1 * inv : (lane == i2) ? w2 * inv : 0.f;
        gates[(size_t)t * 8 + lane] = g;
    }
}

// ---------------- elementwise helpers ----------------
__global__ void gelu_bf16_kernel(const float* __restrict__ x, unsigned short* __restrict__ y, int n) {
    int i = blockIdx.x * 256 + threadIdx.x;
    if (i < n) { float v = x[i]; y[i] = f2bf(0.5f * v * (1.f + erff(v * 0.70710678118f))); }
}
__global__ void zero_kernel(float* __restrict__ x, int n) {
    int i = blockIdx.x * 256 + threadIdx.x;
    if (i < n) x[i] = 0.f;
}
__global__ void add3_kernel(const float* __restrict__ a, const float* __restrict__ b,
                            const float* __restrict__ c, float* __restrict__ o, int n) {
    int i = blockIdx.x * 256 + threadIdx.x;
    if (i < n) o[i] = a[i] + b[i] + c[i];
}

// ---------------- host orchestration ----------------
extern "C" void kernel_launch(void* const* d_in, const int* in_sizes, int n_in,
                              void* d_out, int out_size, void* d_ws, size_t ws_size,
                              hipStream_t stream)
{
    (void)in_sizes; (void)n_in; (void)out_size; (void)ws_size;
    const float* x       = (const float*)d_in[0];
    const float* ln1_w   = (const float*)d_in[1];
    const float* ln2_w   = (const float*)d_in[2];
    const float* W_dq    = (const float*)d_in[3];
    const float* W_uq    = (const float*)d_in[4];
    const float* q_ln_w  = (const float*)d_in[5];
    const float* q_ln_b  = (const float*)d_in[6];
    const float* W_dkv   = (const float*)d_in[7];
    const float* W_ukv   = (const float*)d_in[8];
    const float* kv_ln_w = (const float*)d_in[9];
    const float* kv_ln_b = (const float*)d_in[10];
    const float* W_o     = (const float*)d_in[11];
    const float* s_fc    = (const float*)d_in[12];
    const float* s_proj  = (const float*)d_in[13];
    const float* e_fc    = (const float*)d_in[14];
    const float* e_proj  = (const float*)d_in[15];
    const float* cent    = (const float*)d_in[16];
    const float* rbias   = (const float*)d_in[17];
    float* out = (float*)d_out;

    const int T = 4096, Dm = 768, QP = 384, KVP = 512, FF = 3072;

    char* p = (char*)d_ws;
    auto carve = [&](size_t bytes) -> void* {
        void* r = (void*)p; p += (bytes + 255) & ~(size_t)255; return r;
    };
    unsigned short* wdq_b   = (unsigned short*)carve((size_t)384 * 768 * 2);
    unsigned short* wuq_b   = (unsigned short*)carve((size_t)768 * 384 * 2);
    unsigned short* wdkv_b  = (unsigned short*)carve((size_t)576 * 768 * 2);
    unsigned short* wukv_b  = (unsigned short*)carve((size_t)1152 * 512 * 2);
    unsigned short* wo_b    = (unsigned short*)carve((size_t)768 * 768 * 2);
    unsigned short* sfc_b   = (unsigned short*)carve((size_t)2 * FF * Dm * 2);
    unsigned short* sproj_b = (unsigned short*)carve((size_t)2 * Dm * FF * 2);
    unsigned short* efc_b   = (unsigned short*)carve((size_t)8 * FF * Dm * 2);
    unsigned short* eproj_b = (unsigned short*)carve((size_t)8 * Dm * FF * 2);
    float* cq        = (float*)carve((size_t)T * 384 * 4);
    float* Qf        = (float*)carve((size_t)T * 768 * 4);
    float* ckvR      = (float*)carve((size_t)T * 768 * 4);
    float* gR        = (float*)carve((size_t)T * 3072 * 4);
    float* x1        = (float*)carve((size_t)T * 768 * 4);
    float* gates     = (float*)carve((size_t)T * 8 * 4);
    unsigned short* hln1b = (unsigned short*)carve((size_t)T * 768 * 2);
    unsigned short* cqb   = (unsigned short*)carve((size_t)T * 384 * 2);
    unsigned short* qb    = (unsigned short*)carve((size_t)T * 768 * 2);
    unsigned short* kvlnb = (unsigned short*)carve((size_t)T * 512 * 2);
    unsigned short* kb    = (unsigned short*)carve((size_t)T * 768 * 2);
    unsigned short* vTb   = (unsigned short*)carve((size_t)T * 768 * 2);
    unsigned short* ob    = (unsigned short*)carve((size_t)T * 768 * 2);
    unsigned short* gb    = (unsigned short*)carve((size_t)T * FF * 2);
    float* ckv = ckvR;  float* routedout = ckvR;
    float* KV  = gR;    float* g = gR;
    float* sharedout = Qf;
    unsigned short* h2b = hln1b;

    auto convKN = [&](const float* in, unsigned short* o_, int K, int N, int Npad) {
        int tot = Npad * K;
        conv_kn2nk_kernel<<<(tot + 255) / 256, 256, 0, stream>>>(in, o_, K, N, Npad);
    };
    auto convNK = [&](const float* in, unsigned short* o_, int tot) {
        conv_nk_kernel<<<(tot + 255) / 256, 256, 0, stream>>>(in, o_, tot);
    };
    auto gemm = [&](const unsigned short* A, const unsigned short* Bt, float* C,
                    const float* res, const float* rs, int rsStride,
                    int M, int N, int K, int mode) {
        dim3 grid((N + 63) / 64, (M + 63) / 64);
        gemm_bf16_wmma<<<grid, 128, 0, stream>>>(A, Bt, C, res, rs, rsStride, M, N, K, mode);
    };

    convKN(W_dq,  wdq_b,  Dm, QP, 384);
    convKN(W_uq,  wuq_b,  QP, Dm, 768);
    convKN(W_dkv, wdkv_b, Dm, KVP + 32, 576);
    convKN(W_ukv, wukv_b, KVP, 1152, 1152);
    convNK(W_o,    wo_b,    Dm * Dm);
    convNK(s_fc,   sfc_b,   2 * FF * Dm);
    convNK(s_proj, sproj_b, 2 * Dm * FF);
    convNK(e_fc,   efc_b,   8 * FF * Dm);
    convNK(e_proj, eproj_b, 8 * Dm * FF);

    ln_bf16_kernel<<<T, 256, 0, stream>>>(x, Dm, ln1_w, nullptr, hln1b, Dm, Dm);
    gemm(hln1b, wdq_b, cq, nullptr, nullptr, 0, T, QP, Dm, 0);
    ln_bf16_kernel<<<T, 256, 0, stream>>>(cq, QP, q_ln_w, q_ln_b, cqb, QP, QP);
    gemm(cqb, wuq_b, Qf, nullptr, nullptr, 0, T, Dm, QP, 0);
    gemm(hln1b, wdkv_b, ckv, nullptr, nullptr, 0, T, KVP + 32, Dm, 0);
    ln_bf16_kernel<<<T, 256, 0, stream>>>(ckv, KVP + 32, kv_ln_w, kv_ln_b, kvlnb, KVP, KVP);
    gemm(kvlnb, wukv_b, KV, nullptr, nullptr, 0, T, 1152, KVP, 0);
    rope_q_kernel<<<T, 768, 0, stream>>>(Qf, qb);
    assemble_kv_kernel<<<T, 768, 0, stream>>>(KV, ckv, kb, vTb);
    attn_kernel<<<dim3(16, 12, 4), 128, 0, stream>>>(qb, kb, vTb, ob);
    gemm(ob, wo_b, x1, x, nullptr, 0, T, Dm, Dm, 1);

    ln_bf16_kernel<<<T, 256, 0, stream>>>(x1, Dm, ln2_w, nullptr, h2b, Dm, Dm);
    zero_kernel<<<(T * Dm + 255) / 256, 256, 0, stream>>>(sharedout, T * Dm);
    for (int ns = 0; ns < 2; ++ns) {
        gemm(h2b, sfc_b + (size_t)ns * FF * Dm, g, nullptr, nullptr, 0, T, FF, Dm, 0);
        gelu_bf16_kernel<<<(T * FF + 255) / 256, 256, 0, stream>>>(g, gb, T * FF);
        gemm(gb, sproj_b + (size_t)ns * Dm * FF, sharedout, nullptr, nullptr, 0, T, Dm, FF, 2);
    }
    route_kernel<<<T / 4, 128, 0, stream>>>(h2b, cent, rbias, gates);
    zero_kernel<<<(T * Dm + 255) / 256, 256, 0, stream>>>(routedout, T * Dm);
    for (int e = 0; e < 8; ++e) {
        gemm(h2b, efc_b + (size_t)e * FF * Dm, g, nullptr, nullptr, 0, T, FF, Dm, 0);
        gelu_bf16_kernel<<<(T * FF + 255) / 256, 256, 0, stream>>>(g, gb, T * FF);
        gemm(gb, eproj_b + (size_t)e * Dm * FF, routedout, nullptr, gates + e, 8, T, Dm, FF, 2);
    }
    add3_kernel<<<(T * Dm + 255) / 256, 256, 0, stream>>>(x1, sharedout, routedout, out, T * Dm);
}
